// TransducerJoint_33552284516605
// MI455X (gfx1250) — matrix operand order repaired and checked by reading, not compiled, not run on hardware
//
#include <hip/hip_runtime.h>

// ---------------------------------------------------------------------------
// RNN-T joint:  logits = tanh(enc@We + be + pred@Wp + bp) @ Wo + bo
// B=4 T=256 U=64 D=512 J=1024 V=1024
// Main GEMM M=65536 N=1024 K=1024 -> 137 GFLOP (compute-bound, f16 WMMA,
// f32 accumulate).  Output 256 MB f32 -> ~11us of HBM floor at 23.3 TB/s;
// weights/activations (<9 MB) are L2-resident (192 MB L2).
// ---------------------------------------------------------------------------

typedef __attribute__((ext_vector_type(16))) _Float16 v16h;
typedef __attribute__((ext_vector_type(8)))  _Float16 h8;
typedef __attribute__((ext_vector_type(8)))  float    v8f;

namespace {
constexpr int Bb = 4, Tt = 256, Uu = 64, Dd = 512, Jj = 1024, Vv = 1024;
constexpr int BT   = Bb * Tt;        // 1024 enc rows
constexpr int BU   = Bb * Uu;        // 256  pred rows
constexpr int Mtot = Bb * Tt * Uu;   // 65536 joint rows

constexpr int TM = 128;   // tile rows per workgroup
constexpr int TN = 128;   // tile cols per workgroup (8 waves x 16 rows x 128 cols)
constexpr int TK = 32;    // k-chunk (WMMA f16 K)
// LDS row pitch: 40 halves = 80 B.  80 mod 256 cycles through all sixteen
// 16B-aligned bank offsets -> conflict-free ds_load_b128 fragments, and
// 80 = 5*16 keeps every fragment chunk 16B aligned.
constexpr int PITCH = 40;
} // namespace

// ---- optional CDNA5 async global->LDS path (ASYNCcnt) ----------------------
#if __has_builtin(__builtin_amdgcn_global_load_async_to_lds_b128)
#define USE_ASYNC_LDS 1
#endif

#ifdef USE_ASYNC_LDS
// Builtin signature (from hipcc diagnostic): param0 = V4i in AS1 ("__device__"),
// param1 = V4i in AS3, then two integer immediates (offset, cpol).
typedef int v4i_ __attribute__((vector_size(16)));
typedef __attribute__((address_space(1))) v4i_ v4i_g;
typedef __attribute__((address_space(3))) v4i_ v4i_l;

__device__ __forceinline__ void async_copy16(const _Float16* g, _Float16* l) {
    // AS1 shares representation with generic global pointers (int detour is
    // identity).  Generic LDS pointers are {aperture_hi, lds_offset[31:0]},
    // so truncating to 32 bits recovers the AS3 offset.
    __builtin_amdgcn_global_load_async_to_lds_b128(
        (v4i_g*)(size_t)g,
        (v4i_l*)(size_t)(unsigned)(size_t)l,
        /*offset=*/0, /*cpol=*/0);
}
#endif

__device__ __forceinline__ void async_wait_le2() {
#ifdef USE_ASYNC_LDS
#if __has_builtin(__builtin_amdgcn_s_wait_asynccnt)
    __builtin_amdgcn_s_wait_asynccnt(2);   // allow next chunk's 2 asyncs in flight
#else
    asm volatile("s_wait_asynccnt 0x2" ::: "memory");
#endif
#endif
}
__device__ __forceinline__ void async_wait_le0() {
#ifdef USE_ASYNC_LDS
#if __has_builtin(__builtin_amdgcn_s_wait_asynccnt)
    __builtin_amdgcn_s_wait_asynccnt(0);
#else
    asm volatile("s_wait_asynccnt 0x0" ::: "memory");
#endif
#endif
}

// ---- WMMA fragment loaders -------------------------------------------------
// A fragment (16x32 f16 MxK): lane g=lane/16, m=lane%16; halves [0..15] hold
// K = g*16..g*16+15  -> two ds_load_b128.
__device__ __forceinline__ v16h load_a_frag(const _Float16* As, int rowBase, int lane) {
    const int g = lane >> 4, m = lane & 15;
    const _Float16* p = As + (rowBase + m) * PITCH;
    h8 lo = *(const h8*)(p + g * 16);
    h8 hi = *(const h8*)(p + g * 16 + 8);
    v16h r;
#pragma unroll
    for (int i = 0; i < 8; ++i) { r[i] = lo[i]; r[8 + i] = hi[i]; }
    return r;
}

// B fragment (32x16 f16 KxN): lane n=lane%16, g=lane/16; halves 0..15 hold
// K = g*16..g*16+15 of column n.  B staged transposed [n][k] in LDS.
__device__ __forceinline__ v16h load_b_frag(const _Float16* Bs, int nBase, int lane) {
    const int g = lane >> 4, n = lane & 15;
    const _Float16* p = Bs + (nBase + n) * PITCH + g * 16;
    h8 lo = *(const h8*)(p);
    h8 hi = *(const h8*)(p + 8);
    v16h r;
#pragma unroll
    for (int i = 0; i < 8; ++i) { r[i] = lo[i]; r[8 + i] = hi[i]; }
    return r;
}

__device__ __forceinline__ v8f wmma_f16(v16h a, v16h b, v8f c) {
    return __builtin_amdgcn_wmma_f32_16x16x32_f16(
        false, a, false, b, (short)0, c, false, false);
}

// ---- weight transpose + f32->f16 convert:  W[K,N] -> WT[N,K] ---------------
__global__ __launch_bounds__(256) void cvt_transpose_kernel(
    const float* __restrict__ W, _Float16* __restrict__ WT, int K, int N) {
    int idx = blockIdx.x * 256 + threadIdx.x;
    if (idx >= K * N) return;
    int k = idx / N, n = idx - k * N;
    WT[n * K + k] = (_Float16)W[idx];
}

// ---------------------------------------------------------------------------
// Unified double-buffered WMMA GEMM.
//  FUSED=0:  C[m,:] = A0[m,:] @ WT^T + bias                (projections, K=KDIM)
//  FUSED=1:  C[m,:] = tanh(A0[m/64,:] + A1[b*64+u,:]) @ WT^T + bias  (joint)
// Tile 128x128, 8 waves, each wave 16 rows x 128 cols (8 v8f accumulators).
// Chunk k+1 is staged (A: load+tanh+ds_store, B: async global->LDS) before
// chunk k's 64 WMMAs, hiding memory latency behind matrix math.
// ---------------------------------------------------------------------------
template <int KDIM, bool FUSED>
__global__ __launch_bounds__(256) void gemm_tanh_kernel(
    const float* __restrict__ A0,       // FUSED: encOut[1024,K]  else A[M,K]
    const float* __restrict__ A1,       // FUSED: predOut[256,K]  else unused
    const _Float16* __restrict__ WT,    // [1024, KDIM] f16, transposed
    const float* __restrict__ bias,     // [1024]
    float* __restrict__ C) {            // [M, 1024]
    __shared__ __align__(16) _Float16 As[2][TM * PITCH];
    __shared__ __align__(16) _Float16 Bs[2][TN * PITCH];

    const int tid  = threadIdx.x;
    const int wave = tid >> 5;
    const int lane = tid & 31;
    const int tileM = blockIdx.y * TM;
    const int tileN = blockIdx.x * TN;

    v8f acc[8] = {};

    // ---- A-fill mapping: thread converts 16 halves (half of one 32-wide row)
    const int arow = tid >> 1;
    const int kh   = (tid & 1) * 16;
    const int m    = tileM + arow;
    const float* eP;
    const float* pP = nullptr;
    if constexpr (FUSED) {
        eP = A0 + (size_t)(m >> 6) * KDIM;                        // b*T + t
        pP = A1 + (size_t)(((m >> 14) << 6) | (m & 63)) * KDIM;   // b*U + u
    } else {
        eP = A0 + (size_t)m * KDIM;
    }

    // ---- B-fill mapping: 512 16B chunks, 2 per thread
    const int bn0 = tid >> 2;            // chunk c = tid      -> n 0..63
    const int bn1 = (256 + tid) >> 2;    // chunk c = 256+tid  -> n 64..127
    const int bk  = (tid & 3) * 8;

    auto fillA = [&](int buf, int k0) {
        h8 lo, hi;
#pragma unroll
        for (int i = 0; i < 4; ++i) {
            float4 e = *(const float4*)(eP + k0 + kh + i * 4);
            float4 p = FUSED ? *(const float4*)(pP + k0 + kh + i * 4)
                             : float4{0.f, 0.f, 0.f, 0.f};
            _Float16 t0, t1, t2, t3;
            if constexpr (FUSED) {
                t0 = (_Float16)tanhf(e.x + p.x);
                t1 = (_Float16)tanhf(e.y + p.y);
                t2 = (_Float16)tanhf(e.z + p.z);
                t3 = (_Float16)tanhf(e.w + p.w);
            } else {
                t0 = (_Float16)e.x; t1 = (_Float16)e.y;
                t2 = (_Float16)e.z; t3 = (_Float16)e.w;
            }
            if (i < 2) { lo[i*4+0]=t0; lo[i*4+1]=t1; lo[i*4+2]=t2; lo[i*4+3]=t3; }
            else       { hi[(i-2)*4+0]=t0; hi[(i-2)*4+1]=t1; hi[(i-2)*4+2]=t2; hi[(i-2)*4+3]=t3; }
        }
        *(h8*)&As[buf][arow * PITCH + kh]     = lo;
        *(h8*)&As[buf][arow * PITCH + kh + 8] = hi;
    };

    auto fillB = [&](int buf, int k0) {
        const _Float16* g0 = WT + (size_t)(tileN + bn0) * KDIM + k0 + bk;
        const _Float16* g1 = WT + (size_t)(tileN + bn1) * KDIM + k0 + bk;
        _Float16* l0 = &Bs[buf][bn0 * PITCH + bk];
        _Float16* l1 = &Bs[buf][bn1 * PITCH + bk];
#ifdef USE_ASYNC_LDS
        async_copy16(g0, l0);
        async_copy16(g1, l1);
#else
        *(h8*)l0 = *(const h8*)g0;
        *(h8*)l1 = *(const h8*)g1;
#endif
    };

    auto compute = [&](int buf) {
        v16h a = load_a_frag(&As[buf][0], wave * 16, lane);
#pragma unroll
        for (int s = 0; s < 8; ++s) {
            v16h b = load_b_frag(&Bs[buf][0], s * 16, lane);
            acc[s] = wmma_f16(a, b, acc[s]);
        }
    };

    constexpr int NK = KDIM / TK;
    fillA(0, 0);
    fillB(0, 0);
    for (int kc = 0; kc < NK; ++kc) {
        const int cur = kc & 1;
        const bool more = (kc + 1) < NK;
        if (more) {                       // stage next chunk first
            fillA(1 - cur, (kc + 1) * TK);
            fillB(1 - cur, (kc + 1) * TK);
        }
        // async loads complete in order: waiting <=2 guarantees `cur`'s tile
        // landed while `next`'s two stay in flight.
        if (more) async_wait_le2(); else async_wait_le0();
        __syncthreads();                  // A-tile ds stores + join waves
        compute(cur);                     // 2 + 16 ds_load_b128, 8 v_wmma
        __syncthreads();                  // done reading before overwrite
    }

    // ---- epilogue: C/D layout lane n=lane%16, g=lane/16; VGPR r -> row g*8+r
    const int g = lane >> 4, n0 = lane & 15;
#pragma unroll
    for (int s = 0; s < 8; ++s) {
        const int col = tileN + s * 16 + n0;
        const float bv = bias[col];
#pragma unroll
        for (int r = 0; r < 8; ++r) {
            const int row = tileM + wave * 16 + g * 8 + r;
            C[(size_t)row * Vv + col] = acc[s][r] + bv;
        }
    }
}

// ---------------------------------------------------------------------------
extern "C" void kernel_launch(void* const* d_in, const int* in_sizes, int n_in,
                              void* d_out, int out_size, void* d_ws, size_t ws_size,
                              hipStream_t stream) {
    const float* enc    = (const float*)d_in[0];  // [4,256,512]
    const float* pred   = (const float*)d_in[1];  // [4,64,512]
    const float* W_enc  = (const float*)d_in[2];  // [512,1024]
    const float* b_enc  = (const float*)d_in[3];  // [1024]
    const float* W_pred = (const float*)d_in[4];  // [512,1024]
    const float* b_pred = (const float*)d_in[5];  // [1024]
    const float* W_out  = (const float*)d_in[6];  // [1024,1024]
    const float* b_out  = (const float*)d_in[7];  // [1024]
    float* out = (float*)d_out;                   // [4,256,64,1024]

    // workspace layout (9 MB)
    char* ws = (char*)d_ws;
    _Float16* WencT   = (_Float16*)(ws + (0u << 20));  // [1024,512]  1 MB
    _Float16* WpredT  = (_Float16*)(ws + (1u << 20));  // [1024,512]  1 MB
    _Float16* WoutT   = (_Float16*)(ws + (2u << 20));  // [1024,1024] 2 MB
    float*    encOut  = (float*)   (ws + (4u << 20));  // [1024,1024] 4 MB
    float*    predOut = (float*)   (ws + (8u << 20));  // [256,1024]  1 MB

    // 1) weight transpose + f16 convert (one-time, ~2M elements)
    cvt_transpose_kernel<<<(Dd * Jj) / 256, 256, 0, stream>>>(W_enc,  WencT,  Dd, Jj);
    cvt_transpose_kernel<<<(Dd * Jj) / 256, 256, 0, stream>>>(W_pred, WpredT, Dd, Jj);
    cvt_transpose_kernel<<<(Jj * Vv) / 256, 256, 0, stream>>>(W_out,  WoutT,  Jj, Vv);

    // 2) projections (pre-activation, f32 out)
    gemm_tanh_kernel<Dd, false><<<dim3(Jj / TN, BT / TM), 256, 0, stream>>>(
        enc, nullptr, WencT, b_enc, encOut);
    gemm_tanh_kernel<Dd, false><<<dim3(Jj / TN, BU / TM), 256, 0, stream>>>(
        pred, nullptr, WpredT, b_pred, predOut);

    // 3) fused tanh(enc+pred) @ W_out + b_out  (137 GFLOP)
    gemm_tanh_kernel<Jj, true><<<dim3(Vv / TN, Mtot / TM), 256, 0, stream>>>(
        encOut, predOut, WoutT, b_out, out);
}
